// ConvLSTMClassifier_29248727286182
// MI455X (gfx1250) — matrix operand order repaired and checked by reading, not compiled, not run on hardware
//
#include <hip/hip_runtime.h>
#include <hip/hip_bf16.h>

// ---------------------------------------------------------------------------
// ConvLSTM classifier for MI455X (gfx1250), bf16 WMMA formulation.
// B=64, CH=128, H=128(time), W=512, NC=10.
//
// gates[oc,w] = conv_b[oc] + sum_kx w0[oc][kx]*x[t][w+kx-1]      (VALU)
//             + sum_kx sum_ic Wh[kx][oc][ic]*h[ic][w+kx-1]       (WMMA, K=128)
//
// 256 threads = 8 waves (2 waves/SIMD). Register blocking: 8 w-tiles x 4 gates
// of f32 accumulators (256 VGPRs) stay resident; weight A-fragments and LDS
// B-fragments are software-pipelined one step ahead. h ping-pongs between two
// LDS buffers; c lives in workspace (L2 resident).
// ---------------------------------------------------------------------------

typedef __attribute__((ext_vector_type(16))) __bf16        v16bf;
typedef __attribute__((ext_vector_type(8)))  float          v8f;
typedef __attribute__((ext_vector_type(8)))  unsigned short u16x8;

#define CH   128
#define HH   128
#define WW   512
#define NCLS 10
#define NTHR 256

#define RSTRIDE 136                   // shorts per LDS row; 272B -> conflict-free row access
#define NROWS   514
#define BUFSH   (NROWS * RSTRIDE)     // 69904 shorts per buffer
#define OFF_BUF1   (BUFSH * 2)                  // bytes: 139808
#define OFF_XROW   (OFF_BUF1 * 2)               // 279616
#define OFF_W0     (OFF_XROW + 2 * NROWS * 4)   // 283728
#define OFF_POOL   (OFF_W0 + 512 * 3 * 4)       // 289872
#define LDS_BYTES  (OFF_POOL + 128 * 4)         // 290384

#define WB_BYTES   (3 * 512 * 128 * 2)          // 393216 : packed bf16 A-weights

static __device__ __forceinline__ unsigned short f2bf(float f) {
    union { float f; unsigned u; } a; a.f = f;
    unsigned r = a.u + 0x7FFFu + ((a.u >> 16) & 1u);   // round-to-nearest-even
    return (unsigned short)(r >> 16);
}
static __device__ __forceinline__ float bf2f(unsigned short s) {
    union { unsigned u; float f; } a; a.u = ((unsigned)s) << 16;
    return a.f;
}
static __device__ __forceinline__ v16bf mk16(u16x8 lo, u16x8 hi) {
    union { u16x8 h[2]; v16bf v; } u; u.h[0] = lo; u.h[1] = hi; return u.v;
}
static __device__ __forceinline__ float sigm(float x) { return 1.0f / (1.0f + __expf(-x)); }
static __device__ __forceinline__ float ftanh(float x) { return 2.0f / (1.0f + __expf(-2.0f * x)) - 1.0f; }

// ---------------------------------------------------------------------------
// Pack conv weights (h-channels only, middle kernel row) as bf16:
//   Wb[kx*512*128 + oc*128 + ic] = conv_w[oc][1+ic][1][kx]
// conv_w layout: [512][129][3][3]
// ---------------------------------------------------------------------------
__global__ void convlstm_prep_weights(const float* __restrict__ conv_w,
                                      unsigned short* __restrict__ Wb) {
    int i = blockIdx.x * 256 + threadIdx.x;            // 0 .. 196607
    if (i >= 3 * 512 * 128) return;
    int ic = i & 127;
    int oc = (i >> 7) & 511;
    int kx = i >> 16;
    float v = conv_w[((oc * 129 + (1 + ic)) * 3 + 1) * 3 + kx];
    Wb[i] = f2bf(v);
}

// ---------------------------------------------------------------------------
// Main kernel: one workgroup per batch element, 256 threads (8 waves).
// Wave w owns channel tile [16w, 16w+16) and all 32 w-tiles.
// ---------------------------------------------------------------------------
__global__ __launch_bounds__(NTHR, 1)
void convlstm_kernel(const float* __restrict__ x,
                     const float* __restrict__ conv_w,
                     const float* __restrict__ conv_b,
                     const float* __restrict__ fc_w,
                     const float* __restrict__ fc_b,
                     const unsigned short* __restrict__ Wb,
                     float* __restrict__ c_ws,
                     float* __restrict__ out) {
    extern __shared__ unsigned char smem[];
    unsigned short* buf0   = (unsigned short*)smem;
    float*          xrow   = (float*)(smem + OFF_XROW);   // [2][514], halo at 0 and 513
    float*          w0     = (float*)(smem + OFF_W0);     // [512][3] x-channel taps
    float*          pooled = (float*)(smem + OFF_POOL);   // [128]

    const int b    = blockIdx.x;
    const int tid  = threadIdx.x;
    const int lane = tid & 31;
    const int chtile = tid >> 5;      // wave id = 16-channel tile (0..7)
    const int lh   = lane >> 4;       // lane half (0/1)
    const int ln   = lane & 15;

    float*       cst = c_ws + (size_t)b * (CH * WW);
    const float* xb  = x + (size_t)b * (HH * WW);

    // -------------------- init: zero both h buffers, xrow, c; fill w0, x row 0
    for (int i = tid; i < 2 * BUFSH; i += NTHR) buf0[i] = 0;
    for (int i = tid; i < 2 * NROWS; i += NTHR) {
        float v = 0.0f;
        if (i >= 1 && i <= WW) v = xb[i - 1];            // x row 0 into buffer 0
        xrow[i] = v;
    }
    for (int i = tid; i < 512 * 3; i += NTHR) {
        int oc = i / 3, kx = i % 3;
        w0[i] = conv_w[oc * 1161 + 3 + kx];              // [oc][0][1][kx]
    }
    for (int i = tid; i < CH * WW; i += NTHR) cst[i] = 0.0f;
    __syncthreads();

    // per-lane bias registers: bias[gi][g] = conv_b[gi*128 + chtile*16 + lh*8 + g]
    float bias[4][8];
#pragma unroll
    for (int gi = 0; gi < 4; ++gi)
#pragma unroll
        for (int g = 0; g < 8; ++g)
            bias[gi][g] = conv_b[gi * 128 + chtile * 16 + lh * 8 + g];

    // per-lane A-fragment element offset (shorts) within a (kx,gi) weight block
    const int aoff = (chtile * 16 + ln) * 128 + lh * 8;

    // -------------------- time recurrence
#pragma unroll 1
    for (int t = 0; t < HH; ++t) {
        const unsigned short* src = buf0 + ((t & 1) ? BUFSH : 0);
        unsigned short*       dst = buf0 + ((t & 1) ? 0 : BUFSH);
        const float*          xr  = xrow + (t & 1) * NROWS;

#pragma unroll 1
        for (int wg = 0; wg < 4; ++wg) {                 // 8 w-tiles per group
            const int wbase = wg * 128;                  // first w of this group

            // opaque copy of the weight base: keeps A loads inside this loop
            unsigned long long wbv = (unsigned long long)(uintptr_t)Wb;
            asm volatile("" : "+s"(wbv));
            const unsigned short* wbp = (const unsigned short*)(uintptr_t)wbv;

            // fragment loaders (kk = kx*4 + kc, fully unrolled call sites)
            auto loadA = [&](int kk, v16bf* A) {
                const int kx = kk >> 2, kc = kk & 3;
#pragma unroll
                for (int gi = 0; gi < 4; ++gi) {
                    const unsigned short* a =
                        wbp + (size_t)(kx * 512 + gi * 128) * 128 + aoff + kc * 32;
                    A[gi] = mk16(*(const u16x8*)a, *(const u16x8*)(a + 16));
                }
            };
            auto loadB = [&](int kk, int wt) -> v16bf {
                const int kx = kk >> 2, kc = kk & 3;
                const unsigned short* bq = src + (wbase + kx + ln) * RSTRIDE
                                         + wt * (16 * RSTRIDE) + kc * 32 + lh * 8;
                return mk16(*(const u16x8*)bq, *(const u16x8*)(bq + 16));
            };

            v8f acc[8][4];                               // 256 VGPRs, stays resident
#pragma unroll
            for (int wt = 0; wt < 8; ++wt)
#pragma unroll
                for (int gi = 0; gi < 4; ++gi)
#pragma unroll
                    for (int g = 0; g < 8; ++g) acc[wt][gi][g] = 0.0f;

            // software-pipelined GEMM: A one (kx,kc) ahead, B one w-tile ahead
            v16bf Acur[4], Anext[4];
            loadA(0, Acur);
            v16bf Bcur = loadB(0, 0);
#pragma unroll
            for (int kk = 0; kk < 12; ++kk) {
                if (kk < 11) loadA(kk + 1, Anext);
#pragma unroll
                for (int wt = 0; wt < 8; ++wt) {
                    v16bf Bnext = Bcur;
                    if (wt < 7)       Bnext = loadB(kk, wt + 1);
                    else if (kk < 11) Bnext = loadB(kk + 1, 0);
#pragma unroll
                    for (int gi = 0; gi < 4; ++gi)
                        acc[wt][gi] = __builtin_amdgcn_wmma_f32_16x16x32_bf16(
                            false, Acur[gi], false, Bcur, (short)0, acc[wt][gi],
                            false, false);
                    Bcur = Bnext;
                }
#pragma unroll
                for (int gi = 0; gi < 4; ++gi) Acur[gi] = Anext[gi];
            }

            // pointwise LSTM update (C/D layout: lane ln -> N=w offset, VGPR g -> M)
#pragma unroll
            for (int wt = 0; wt < 8; ++wt) {
                const int w   = wbase + wt * 16 + ln;
                const float xm1 = xr[w];
                const float x0v = xr[w + 1];
                const float xp1 = xr[w + 2];
#pragma unroll
                for (int g = 0; g < 8; ++g) {
                    const int ch = chtile * 16 + lh * 8 + g;
                    float gv[4];
#pragma unroll
                    for (int gi = 0; gi < 4; ++gi) {
                        const float* wp = w0 + (gi * 128 + ch) * 3;
                        gv[gi] = acc[wt][gi][g] + bias[gi][g]
                               + wp[0] * xm1 + wp[1] * x0v + wp[2] * xp1;
                    }
                    const float iv = sigm(gv[0]);
                    const float fv = sigm(gv[1]);
                    const float gg = ftanh(gv[2]);
                    const float ov = sigm(gv[3]);
                    const float cn = fv * cst[ch * WW + w] + iv * gg;
                    cst[ch * WW + w] = cn;
                    dst[(w + 1) * RSTRIDE + ch] = f2bf(ov * ftanh(cn));
                }
            }
        }

        // stage next x row into the other xrow buffer
        if (t + 1 < HH) {
            for (int i = tid; i < WW; i += NTHR)
                xrow[((t + 1) & 1) * NROWS + i + 1] = xb[(t + 1) * WW + i];
        }
        __syncthreads();
    }

    // -------------------- pool + classifier (t=127 wrote buffer 0)
    if (tid < CH) {
        float s = 0.0f;
        for (int w = 0; w < WW; ++w) s += bf2f(buf0[(w + 1) * RSTRIDE + tid]);
        pooled[tid] = s * (1.0f / (float)WW);
    }
    __syncthreads();
    if (tid < NCLS) {
        float s = fc_b[tid];
        for (int ch = 0; ch < CH; ++ch) s += pooled[ch] * fc_w[tid * CH + ch];
        out[b * NCLS + tid] = s;
    }
}

// ---------------------------------------------------------------------------
extern "C" void kernel_launch(void* const* d_in, const int* in_sizes, int n_in,
                              void* d_out, int out_size, void* d_ws, size_t ws_size,
                              hipStream_t stream) {
    const float* x      = (const float*)d_in[0];
    const float* conv_w = (const float*)d_in[1];
    const float* conv_b = (const float*)d_in[2];
    const float* fc_w   = (const float*)d_in[3];
    const float* fc_b   = (const float*)d_in[4];
    float*       out    = (float*)d_out;

    unsigned short* Wb  = (unsigned short*)d_ws;                       // 384 KB
    float*          cws = (float*)((char*)d_ws + WB_BYTES);            // 16 MB c-state

    (void)hipFuncSetAttribute(reinterpret_cast<const void*>(convlstm_kernel),
                              hipFuncAttributeMaxDynamicSharedMemorySize, LDS_BYTES);

    convlstm_prep_weights<<<768, 256, 0, stream>>>(conv_w, Wb);
    convlstm_kernel<<<64, NTHR, LDS_BYTES, stream>>>(x, conv_w, conv_b, fc_w, fc_b,
                                                     Wb, cws, out);
    (void)in_sizes; (void)n_in; (void)out_size; (void)ws_size;
}
